// DualStreamBackbone_38397007626920
// MI455X (gfx1250) — compile-verified, matrix operand order checked by reading, not run on hardware
//
#include <hip/hip_runtime.h>

typedef __bf16 bf16;
typedef bf16  v16bf __attribute__((ext_vector_type(16)));
typedef float v8f   __attribute__((ext_vector_type(8)));

#define B_    256
#define TAU_  256
#define ROWS_ (B_ * TAU_)
#define EPS_  1e-5f
#define KF_   96   // fused K padded 65 -> 96

__device__ __forceinline__ bf16 f2bf(float f) {
  unsigned u = __builtin_bit_cast(unsigned, f);
  u += 0x7FFFu + ((u >> 16) & 1u);              // round-to-nearest-even
  unsigned short s = (unsigned short)(u >> 16);
  return __builtin_bit_cast(bf16, s);
}

// k-pair index for (vgpr v, lane-group g) per CDNA5 16-bit A/B VGPR layout:
// v0..3: g0 -> k pairs 0..3, g1 -> 4..7 ; v4..7: g0 -> 8..11, g1 -> 12..15
__device__ __forceinline__ int pair_of(int v, int g) {
  return (v & 3) + (g << 2) + ((v >> 2) << 3);
}

// A fragment (16x32 bf16) from row-major bf16 [16 rows x ld], k-window [k0,k0+32)
__device__ __forceinline__ v16bf load_a_bf16(const bf16* a, int ld, int k0, int lane) {
  int m = lane & 15, g = lane >> 4;
  v16bf r;
#pragma unroll
  for (int v = 0; v < 8; ++v) {
    int k = k0 + 2 * pair_of(v, g);
    const bf16* s = a + m * ld + k;
    r[2 * v] = s[0];
    r[2 * v + 1] = s[1];
  }
  return r;
}

// A fragment from row-major f32 (convert on load)
__device__ __forceinline__ v16bf load_a_f32(const float* a, int ld, int k0, int lane) {
  int m = lane & 15, g = lane >> 4;
  v16bf r;
#pragma unroll
  for (int v = 0; v < 8; ++v) {
    int k = k0 + 2 * pair_of(v, g);
    const float* s = a + m * ld + k;
    r[2 * v] = f2bf(s[0]);
    r[2 * v + 1] = f2bf(s[1]);
  }
  return r;
}

// B fragment from k-pair-interleaved storage: element (k,n) at [(k>>1)*2*N + 2*n + (k&1)]
__device__ __forceinline__ v16bf load_b_pp(const bf16* bp, int N, int k0, int n0, int lane) {
  int n = n0 + (lane & 15), g = lane >> 4;
  v16bf r;
#pragma unroll
  for (int v = 0; v < 8; ++v) {
    int p = (k0 >> 1) + pair_of(v, g);
    const bf16* s = bp + p * 2 * N + 2 * n;
    r[2 * v] = s[0];
    r[2 * v + 1] = s[1];
  }
  return r;
}

#define WMMA_BF16(a, b, c) \
  __builtin_amdgcn_wmma_f32_16x16x32_bf16(false, (a), false, (b), (short)0, (c), false, false)

// ---------------------------------------------------------------------------
// Kernel 1: CNN stems + degenerate cross-attn + LN -> fused bf16 [ROWS x 96]
// ---------------------------------------------------------------------------
__global__ void k_stem(const float* __restrict__ xs_g, const float* __restrict__ xc_g,
                       const float* __restrict__ rho,
                       const float* __restrict__ sw1, const float* __restrict__ sb1,
                       const float* __restrict__ sw2, const float* __restrict__ sb2,
                       const float* __restrict__ s1g, const float* __restrict__ s1b,
                       const float* __restrict__ s1m, const float* __restrict__ s1v,
                       const float* __restrict__ s2g, const float* __restrict__ s2b,
                       const float* __restrict__ s2m, const float* __restrict__ s2v,
                       const float* __restrict__ cw1, const float* __restrict__ cb1,
                       const float* __restrict__ cw2, const float* __restrict__ cb2,
                       const float* __restrict__ c1g, const float* __restrict__ c1b,
                       const float* __restrict__ c1m, const float* __restrict__ c1v,
                       const float* __restrict__ c2g, const float* __restrict__ c2b,
                       const float* __restrict__ c2m, const float* __restrict__ c2v,
                       const float* __restrict__ wv, const float* __restrict__ bv,
                       const float* __restrict__ wo, const float* __restrict__ bo,
                       const float* __restrict__ lng, const float* __restrict__ lnb,
                       bf16* __restrict__ fused) {
  __shared__ float W1s[64][32], W1c[64][32];
  __shared__ float W2s[32][32], W2c[32][32], WVt[32][32], WOt[32][32];
  __shared__ float sc1s[32], sh1s[32], sc2s[32], sh2s[32];
  __shared__ float sc1c[32], sh1c[32], sc2c[32], sh2c[32];
  __shared__ float Bv[32], Bo[32], B1s[32], B2s[32], B1c[32], B2c[32], G[32], Bt[32];
  int tid = threadIdx.x;
  for (int i = tid; i < 64 * 32; i += 256) {
    int d = i >> 5, c = i & 31;
    W1s[d][c] = sw1[c * 192 + d * 3 + 1];   // center tap of conv k=3
    W1c[d][c] = cw1[c * 192 + d * 3 + 1];
  }
  for (int i = tid; i < 32 * 32; i += 256) {
    int d = i >> 5, c = i & 31;
    W2s[d][c] = sw2[c * 96 + d * 3 + 1];
    W2c[d][c] = cw2[c * 96 + d * 3 + 1];
    WVt[d][c] = wv[c * 32 + d];
    WOt[d][c] = wo[c * 32 + d];
  }
  if (tid < 32) {
    int c = tid;
    sc1s[c] = s1g[c] * rsqrtf(s1v[c] + EPS_); sh1s[c] = s1b[c] - s1m[c] * sc1s[c];
    sc2s[c] = s2g[c] * rsqrtf(s2v[c] + EPS_); sh2s[c] = s2b[c] - s2m[c] * sc2s[c];
    sc1c[c] = c1g[c] * rsqrtf(c1v[c] + EPS_); sh1c[c] = c1b[c] - c1m[c] * sc1c[c];
    sc2c[c] = c2g[c] * rsqrtf(c2v[c] + EPS_); sh2c[c] = c2b[c] - c2m[c] * sc2c[c];
    Bv[c] = bv[c]; Bo[c] = bo[c];
    B1s[c] = sb1[c]; B2s[c] = sb2[c]; B1c[c] = cb1[c]; B2c[c] = cb2[c];
    G[c] = lng[c]; Bt[c] = lnb[c];
  }
  __syncthreads();

  int r = blockIdx.x * 256 + tid;

  float cf[32];
  {
    float acc[32];
#pragma unroll
    for (int c = 0; c < 32; ++c) acc[c] = B1c[c];
    for (int d = 0; d < 64; ++d) {
      float x = xc_g[(size_t)r * 64 + d];
#pragma unroll
      for (int c = 0; c < 32; ++c) acc[c] += W1c[d][c] * x;
    }
    float h1[32];
#pragma unroll
    for (int c = 0; c < 32; ++c) { float t = acc[c] * sc1c[c] + sh1c[c]; h1[c] = t > 0.f ? t : 0.f; }
#pragma unroll
    for (int c = 0; c < 32; ++c) acc[c] = B2c[c];
    for (int d = 0; d < 32; ++d) {
      float x = h1[d];
#pragma unroll
      for (int c = 0; c < 32; ++c) acc[c] += W2c[d][c] * x;
    }
#pragma unroll
    for (int c = 0; c < 32; ++c) { float t = acc[c] * sc2c[c] + sh2c[c]; cf[c] = t > 0.f ? t : 0.f; }
  }
  float sf[32];
  {
    float acc[32];
#pragma unroll
    for (int c = 0; c < 32; ++c) acc[c] = B1s[c];
    for (int d = 0; d < 64; ++d) {
      float x = xs_g[(size_t)r * 64 + d];
#pragma unroll
      for (int c = 0; c < 32; ++c) acc[c] += W1s[d][c] * x;
    }
    float h1[32];
#pragma unroll
    for (int c = 0; c < 32; ++c) { float t = acc[c] * sc1s[c] + sh1s[c]; h1[c] = t > 0.f ? t : 0.f; }
#pragma unroll
    for (int c = 0; c < 32; ++c) acc[c] = B2s[c];
    for (int d = 0; d < 32; ++d) {
      float x = h1[d];
#pragma unroll
      for (int c = 0; c < 32; ++c) acc[c] += W2s[d][c] * x;
    }
#pragma unroll
    for (int c = 0; c < 32; ++c) { float t = acc[c] * sc2s[c] + sh2s[c]; sf[c] = t > 0.f ? t : 0.f; }
  }
  // cross-attn (softmax over one key == 1): out = Wo (Wv c + bv) + bo
  float v[32];
#pragma unroll
  for (int o = 0; o < 32; ++o) v[o] = Bv[o];
  for (int i = 0; i < 32; ++i) {
    float x = cf[i];
#pragma unroll
    for (int o = 0; o < 32; ++o) v[o] += WVt[i][o] * x;
  }
  float at[32];
#pragma unroll
  for (int o = 0; o < 32; ++o) at[o] = Bo[o];
  for (int i = 0; i < 32; ++i) {
    float x = v[i];
#pragma unroll
    for (int o = 0; o < 32; ++o) at[o] += WOt[i][o] * x;
  }
  float y[32], mu = 0.f, s2 = 0.f;
#pragma unroll
  for (int c = 0; c < 32; ++c) { y[c] = sf[c] + at[c]; mu += y[c]; }
  mu *= (1.f / 32.f);
#pragma unroll
  for (int c = 0; c < 32; ++c) { float d = y[c] - mu; s2 += d * d; }
  s2 *= (1.f / 32.f);
  float inv = rsqrtf(s2 + EPS_);

  bf16* fr = fused + (size_t)r * KF_;
#pragma unroll
  for (int c = 0; c < 32; ++c) fr[c] = f2bf(G[c] * (y[c] - mu) * inv + Bt[c]);
#pragma unroll
  for (int c = 0; c < 32; ++c) fr[32 + c] = f2bf(cf[c]);
  fr[64] = f2bf(rho[r]);
#pragma unroll
  for (int c = 65; c < 96; ++c) fr[c] = f2bf(0.f);
}

// ---------------------------------------------------------------------------
// Kernel 2: x_proj = fused @ Wih^T + (bih+bhh)  [ROWS x 512], K=96.
// 2 M-tiles per wave -> every B-fragment gather feeds 2 WMMAs.
// ---------------------------------------------------------------------------
__global__ void k_xproj(const bf16* __restrict__ fused, const float* __restrict__ wih,
                        const float* __restrict__ bih, const float* __restrict__ bhh,
                        float* __restrict__ xproj) {
  __shared__ bf16 Bp[48 * 2 * 512];  // 96 KB, k-pair interleaved, N=512
  __shared__ float bias[512];
  int tid = threadIdx.x;
  for (int i = tid; i < 96 * 512; i += 256) {
    int k = i >> 9, n = i & 511;
    float val = (k < 65) ? wih[n * 65 + k] : 0.f;
    Bp[(k >> 1) * 1024 + 2 * n + (k & 1)] = f2bf(val);
  }
  for (int i = tid; i < 512; i += 256) bias[i] = bih[i] + bhh[i];
  __syncthreads();

  int lane = tid & 31, w = tid >> 5;
  int g = lane >> 4, nl = lane & 15;
  size_t r0 = (size_t)blockIdx.x * 256 + (size_t)w * 16;  // tile0
  size_t r1 = r0 + 128;                                   // tile1
  v16bf a0[3], a1[3];
#pragma unroll
  for (int ks = 0; ks < 3; ++ks) {
    a0[ks] = load_a_bf16(fused + r0 * KF_, KF_, ks * 32, lane);
    a1[ks] = load_a_bf16(fused + r1 * KF_, KF_, ks * 32, lane);
  }
  for (int nt = 0; nt < 32; ++nt) {
    v16bf b0 = load_b_pp(Bp, 512, 0, nt * 16, lane);
    v16bf b1 = load_b_pp(Bp, 512, 32, nt * 16, lane);
    v16bf b2 = load_b_pp(Bp, 512, 64, nt * 16, lane);
    v8f acc0 = {}, acc1 = {};
    acc0 = WMMA_BF16(a0[0], b0, acc0);
    acc1 = WMMA_BF16(a1[0], b0, acc1);
    acc0 = WMMA_BF16(a0[1], b1, acc0);
    acc1 = WMMA_BF16(a1[1], b1, acc1);
    acc0 = WMMA_BF16(a0[2], b2, acc0);
    acc1 = WMMA_BF16(a1[2], b2, acc1);
    int n = nt * 16 + nl;
    float bsn = bias[n];
#pragma unroll
    for (int rr = 0; rr < 8; ++rr) {
      int m = rr + 8 * g;
      xproj[(r0 + m) * 512 + n] = acc0[rr] + bsn;
      xproj[(r1 + m) * 512 + n] = acc1[rr] + bsn;
    }
  }
}

// ---------------------------------------------------------------------------
// Kernel 3: persistent LSTM. 16 blocks x 16 batch rows. Whh B-fragments are
// hoisted into registers once (loop-invariant) -> zero B-side LDS traffic in
// the 256-step recurrence. x_proj gate loads issued early + prefetch of t+1.
// ---------------------------------------------------------------------------
__global__ void k_lstm(const float* __restrict__ xproj, const float* __restrict__ whh,
                       float* __restrict__ lstm_out) {
  __shared__ bf16 Wpp[64 * 2 * 512];  // 128 KB: Whh^T k-pair interleaved (K=128, N=512)
  __shared__ bf16 Hs[16 * 128];       // h state bf16
  __shared__ float Cs[16 * 128];      // c state f32
  int tid = threadIdx.x;
  for (int i = tid; i < 128 * 512; i += 256) {
    int k = i >> 9, n = i & 511;
    Wpp[(k >> 1) * 1024 + 2 * n + (k & 1)] = f2bf(whh[n * 128 + k]);
  }
  for (int i = tid; i < 16 * 128; i += 256) { Hs[i] = f2bf(0.f); Cs[i] = 0.f; }
  __syncthreads();

  int lane = tid & 31, w = tid >> 5;
  int g = lane >> 4, nl = lane & 15;
  int hidx = w * 16 + nl;
  size_t b0 = (size_t)blockIdx.x * 16;

  // loop-invariant Whh fragments: [gate][kslab]
  v16bf bfr[4][4];
#pragma unroll
  for (int gate = 0; gate < 4; ++gate)
#pragma unroll
    for (int ks = 0; ks < 4; ++ks)
      bfr[gate][ks] = load_b_pp(Wpp, 512, ks * 32, gate * 128 + w * 16, lane);

  for (int t = 0; t < TAU_; ++t) {
    // issue gate-input loads early so HBM latency overlaps the WMMAs
    float xp[4][8];
#pragma unroll
    for (int rr = 0; rr < 8; ++rr) {
      int m = rr + 8 * g;
      size_t rowbase = ((b0 + m) * TAU_ + t) * 512 + hidx;
#pragma unroll
      for (int gate = 0; gate < 4; ++gate) xp[gate][rr] = xproj[rowbase + gate * 128];
      if (t + 1 < TAU_) __builtin_prefetch(&xproj[rowbase + 512], 0, 1);
    }

    v16bf a[4];
#pragma unroll
    for (int ks = 0; ks < 4; ++ks) a[ks] = load_a_bf16(Hs, 128, ks * 32, lane);
    __syncthreads();  // all reads of h done before updates

    v8f acc[4];
#pragma unroll
    for (int gate = 0; gate < 4; ++gate) {
      v8f c = {};
#pragma unroll
      for (int ks = 0; ks < 4; ++ks) c = WMMA_BF16(a[ks], bfr[gate][ks], c);
      acc[gate] = c;
    }
#pragma unroll
    for (int rr = 0; rr < 8; ++rr) {
      int m = rr + 8 * g;
      float gi = acc[0][rr] + xp[0][rr];
      float gf = acc[1][rr] + xp[1][rr];
      float gg = acc[2][rr] + xp[2][rr];
      float go = acc[3][rr] + xp[3][rr];
      float si = 1.f / (1.f + __expf(-gi));
      float sf = 1.f / (1.f + __expf(-gf));
      float so = 1.f / (1.f + __expf(-go));
      float cn = sf * Cs[m * 128 + hidx] + si * tanhf(gg);
      Cs[m * 128 + hidx] = cn;
      float h = so * tanhf(cn);
      Hs[m * 128 + hidx] = f2bf(h);
      lstm_out[((b0 + m) * TAU_ + t) * 128 + hidx] = h;
    }
    __syncthreads();  // updates visible before next step's reads
  }
}

// ---------------------------------------------------------------------------
// Kernel 4: qkv = lstm_out @ Win^T + bin -> bf16 [ROWS x 384]; 2 M-tiles/wave
// ---------------------------------------------------------------------------
__global__ void k_qkv(const float* __restrict__ lstm_out, const float* __restrict__ win,
                      const float* __restrict__ bin, bf16* __restrict__ qkv) {
  __shared__ bf16 Bp[64 * 2 * 384];  // 96 KB
  __shared__ float bias[384];
  int tid = threadIdx.x;
  for (int i = tid; i < 128 * 384; i += 256) {
    int k = i / 384, n = i % 384;
    Bp[(k >> 1) * 768 + 2 * n + (k & 1)] = f2bf(win[n * 128 + k]);
  }
  for (int i = tid; i < 384; i += 256) bias[i] = bin[i];
  __syncthreads();

  int lane = tid & 31, w = tid >> 5;
  int g = lane >> 4, nl = lane & 15;
  size_t r0 = (size_t)blockIdx.x * 256 + (size_t)w * 16;
  size_t r1 = r0 + 128;
  v16bf a0[4], a1[4];
#pragma unroll
  for (int ks = 0; ks < 4; ++ks) {
    a0[ks] = load_a_f32(lstm_out + r0 * 128, 128, ks * 32, lane);
    a1[ks] = load_a_f32(lstm_out + r1 * 128, 128, ks * 32, lane);
  }
  for (int nt = 0; nt < 24; ++nt) {
    v8f acc0 = {}, acc1 = {};
#pragma unroll
    for (int ks = 0; ks < 4; ++ks) {
      v16bf b = load_b_pp(Bp, 384, ks * 32, nt * 16, lane);
      acc0 = WMMA_BF16(a0[ks], b, acc0);
      acc1 = WMMA_BF16(a1[ks], b, acc1);
    }
    int n = nt * 16 + nl;
    float bs = bias[n];
#pragma unroll
    for (int rr = 0; rr < 8; ++rr) {
      qkv[(r0 + rr + 8 * g) * 384 + n] = f2bf(acc0[rr] + bs);
      qkv[(r1 + rr + 8 * g) * 384 + n] = f2bf(acc1[rr] + bs);
    }
  }
}

// ---------------------------------------------------------------------------
// Kernel 5: attention per (b, head): S=q k^T (K=32), softmax, ctx=P V
// ---------------------------------------------------------------------------
__global__ void k_attn(const bf16* __restrict__ qkv, float* __restrict__ ctx) {
  __shared__ bf16 Kp[16 * 2 * 256];   // 16 KB  (K-dim=d, N=key)
  __shared__ bf16 Vp[128 * 2 * 32];   // 16 KB  (K-dim=key, N=d)
  __shared__ bf16 P[8][16 * 256];     // 64 KB  per-wave probability tiles
  int tid = threadIdx.x;
  int b = blockIdx.x >> 2, h = blockIdx.x & 3;
  const bf16* base = qkv + (size_t)b * TAU_ * 384;
  for (int i = tid; i < 32 * 256; i += 256) {
    int d = i >> 8, key = i & 255;
    Kp[(d >> 1) * 512 + 2 * key + (d & 1)] = base[key * 384 + 128 + h * 32 + d];
  }
  for (int i = tid; i < 256 * 32; i += 256) {
    int key = i >> 5, d = i & 31;
    Vp[(key >> 1) * 64 + 2 * d + (key & 1)] = base[key * 384 + 256 + h * 32 + d];
  }
  __syncthreads();

  int lane = tid & 31, w = tid >> 5;
  int g = lane >> 4, nl = lane & 15;
  const float scale = 0.17677669529663687f;  // 1/sqrt(32)
  bf16* Pw = &P[w][0];

  for (int qi = 0; qi < 2; ++qi) {
    int qt = w + 8 * qi;
    const bf16* Aq = base + (size_t)(qt * 16) * 384 + h * 32;
    v16bf aq = load_a_bf16(Aq, 384, 0, lane);
    float S[16][8];
#pragma unroll
    for (int kt = 0; kt < 16; ++kt) {
      v8f c = {};
      c = WMMA_BF16(aq, load_b_pp(Kp, 256, 0, kt * 16, lane), c);
#pragma unroll
      for (int rr = 0; rr < 8; ++rr) S[kt][rr] = c[rr] * scale;
    }
#pragma unroll
    for (int rr = 0; rr < 8; ++rr) {
      float mx = -1e30f;
#pragma unroll
      for (int kt = 0; kt < 16; ++kt) mx = fmaxf(mx, S[kt][rr]);
      for (int mask = 1; mask < 16; mask <<= 1) mx = fmaxf(mx, __shfl_xor(mx, mask, 32));
      float e[16], sum = 0.f;
#pragma unroll
      for (int kt = 0; kt < 16; ++kt) { e[kt] = __expf(S[kt][rr] - mx); sum += e[kt]; }
      for (int mask = 1; mask < 16; mask <<= 1) sum += __shfl_xor(sum, mask, 32);
      float inv = 1.f / sum;
      int m = rr + 8 * g;
#pragma unroll
      for (int kt = 0; kt < 16; ++kt) Pw[m * 256 + kt * 16 + nl] = f2bf(e[kt] * inv);
    }
#pragma unroll
    for (int nt = 0; nt < 2; ++nt) {
      v8f c = {};
#pragma unroll
      for (int ks = 0; ks < 8; ++ks)
        c = WMMA_BF16(load_a_bf16(Pw, 256, ks * 32, lane),
                      load_b_pp(Vp, 32, ks * 32, nt * 16, lane), c);
      int dcol = nt * 16 + nl;
#pragma unroll
      for (int rr = 0; rr < 8; ++rr) {
        int trow = qt * 16 + rr + 8 * g;
        ctx[((size_t)b * TAU_ + trow) * 128 + h * 32 + dcol] = c[rr];
      }
    }
  }
}

// ---------------------------------------------------------------------------
// Kernel 6: attn_out = ctx @ Wo^T + bo; LN(lstm_out + attn_out); mean over tau
// Both M-tiles of a wave processed together: B gathers amortized over 2 WMMAs.
// ---------------------------------------------------------------------------
__global__ void k_final(const float* __restrict__ ctx, const float* __restrict__ lstm_out,
                        const float* __restrict__ wo, const float* __restrict__ bo,
                        const float* __restrict__ lng, const float* __restrict__ lnb,
                        float* __restrict__ out) {
  __shared__ bf16 Wp[64 * 2 * 128];      // 32 KB
  __shared__ float rowbuf[8][32 * 128];  // 128 KB per-wave tiles (2 M-tiles)
  __shared__ float accb[8][128];         // per-wave mean accumulators
  __shared__ float bias[128], gln[128], bln[128];
  int tid = threadIdx.x;
  for (int i = tid; i < 128 * 128; i += 256) {
    int k = i >> 7, n = i & 127;
    Wp[(k >> 1) * 256 + 2 * n + (k & 1)] = f2bf(wo[n * 128 + k]);
  }
  if (tid < 128) { bias[tid] = bo[tid]; gln[tid] = lng[tid]; bln[tid] = lnb[tid]; }
  for (int i = tid; i < 1024; i += 256) accb[i >> 7][i & 127] = 0.f;
  __syncthreads();

  int lane = tid & 31, w = tid >> 5;
  int g = lane >> 4, nl = lane & 15;
  size_t b = blockIdx.x;
  float* rb = &rowbuf[w][0];
  float* acw = &accb[w][0];

  size_t t0 = b * TAU_ + (size_t)w * 16;         // tile0 rows
  size_t t1 = b * TAU_ + (size_t)(w + 8) * 16;   // tile1 rows
  v16bf a0[4], a1[4];
#pragma unroll
  for (int ks = 0; ks < 4; ++ks) {
    a0[ks] = load_a_f32(ctx + t0 * 128, 128, ks * 32, lane);
    a1[ks] = load_a_f32(ctx + t1 * 128, 128, ks * 32, lane);
  }
#pragma unroll
  for (int nt = 0; nt < 8; ++nt) {
    v8f c0 = {}, c1 = {};
#pragma unroll
    for (int ks = 0; ks < 4; ++ks) {
      v16bf bfrag = load_b_pp(Wp, 128, ks * 32, nt * 16, lane);
      c0 = WMMA_BF16(a0[ks], bfrag, c0);
      c1 = WMMA_BF16(a1[ks], bfrag, c1);
    }
    int n = nt * 16 + nl;
#pragma unroll
    for (int rr = 0; rr < 8; ++rr) {
      int m = rr + 8 * g;
      rb[m * 128 + n] = c0[rr] + bias[n] + lstm_out[(t0 + m) * 128 + n];
      rb[(16 + m) * 128 + n] = c1[rr] + bias[n] + lstm_out[(t1 + m) * 128 + n];
    }
  }
  __syncthreads();
  for (int rrow = 0; rrow < 32; ++rrow) {
    int c0 = lane * 4;
    float x0 = rb[rrow * 128 + c0 + 0], x1 = rb[rrow * 128 + c0 + 1];
    float x2 = rb[rrow * 128 + c0 + 2], x3 = rb[rrow * 128 + c0 + 3];
    float s = x0 + x1 + x2 + x3;
    float q = x0 * x0 + x1 * x1 + x2 * x2 + x3 * x3;
    for (int mask = 1; mask < 32; mask <<= 1) {
      s += __shfl_xor(s, mask, 32);
      q += __shfl_xor(q, mask, 32);
    }
    float mu = s * (1.f / 128.f);
    float var = q * (1.f / 128.f) - mu * mu;
    float inv = rsqrtf(var + EPS_);
    acw[c0 + 0] += gln[c0 + 0] * (x0 - mu) * inv + bln[c0 + 0];
    acw[c0 + 1] += gln[c0 + 1] * (x1 - mu) * inv + bln[c0 + 1];
    acw[c0 + 2] += gln[c0 + 2] * (x2 - mu) * inv + bln[c0 + 2];
    acw[c0 + 3] += gln[c0 + 3] * (x3 - mu) * inv + bln[c0 + 3];
  }
  __syncthreads();
  if (tid < 128) {
    float s = 0.f;
#pragma unroll
    for (int w2 = 0; w2 < 8; ++w2) s += accb[w2][tid];
    out[b * 128 + tid] = s * (1.f / 256.f);
  }
}

// ---------------------------------------------------------------------------
extern "C" void kernel_launch(void* const* d_in, const int* in_sizes, int n_in,
                              void* d_out, int out_size, void* d_ws, size_t ws_size,
                              hipStream_t stream) {
  (void)in_sizes; (void)n_in; (void)out_size; (void)ws_size;
#define FI(i) ((const float*)d_in[(i)])

  char* wsb = (char*)d_ws;
  bf16*  fused  = (bf16*)(wsb + 0);                       //  12,582,912 B
  float* xproj  = (float*)(wsb + 12582912ull);            // 134,217,728 B
  float* lstm_o = (float*)(wsb + 146800640ull);           //  33,554,432 B
  bf16*  qkv    = (bf16*)(wsb + 180355072ull);            //  50,331,648 B
  float* ctxb   = (float*)(wsb + 230686720ull);           //  33,554,432 B

  k_stem<<<ROWS_ / 256, 256, 0, stream>>>(
      FI(0), FI(1), FI(2),
      FI(3), FI(4), FI(5), FI(6), FI(7), FI(8), FI(9), FI(10),
      FI(11), FI(12), FI(13), FI(14),
      FI(15), FI(16), FI(17), FI(18), FI(19), FI(20), FI(21), FI(22),
      FI(23), FI(24), FI(25), FI(26),
      FI(27), FI(28), FI(29), FI(30), FI(31), FI(32),
      fused);

  k_xproj<<<ROWS_ / 256, 256, 0, stream>>>(fused, FI(33), FI(35), FI(36), xproj);

  k_lstm<<<B_ / 16, 256, 0, stream>>>(xproj, FI(34), lstm_o);

  k_qkv<<<ROWS_ / 256, 256, 0, stream>>>(lstm_o, FI(37), FI(38), qkv);

  k_attn<<<B_ * 4, 256, 0, stream>>>(qkv, ctxb);

  k_final<<<B_, 256, 0, stream>>>(ctxb, lstm_o, FI(39), FI(40), FI(41), FI(42),
                                  (float*)d_out);
#undef FI
}
